// DVAE_51960514347228
// MI455X (gfx1250) — compile-verified, matrix operand order checked by reading, not run on hardware
//
#include <hip/hip_runtime.h>
#include <hip/hip_bf16.h>
#include <math.h>

// DVAE forward for MI455X (gfx1250, wave32).
// All GEMMs: v_wmma_f32_16x16x32_f16, f16-resident activations/weights,
// f32 accumulate + f32 epilogue math. Forward RNN of the reference is dead
// code (out_f unused) and is skipped.

typedef __attribute__((ext_vector_type(16))) _Float16 v16h;
typedef __attribute__((ext_vector_type(2)))  _Float16 h2;
typedef __attribute__((ext_vector_type(8)))  float    v8f;

namespace {
constexpr int kB = 128, kS = 256, kIN = 88, kH = 400, kL = 100, kHEM = 100, kHTR = 200;
}

__device__ __forceinline__ v8f wmma_f16(v16h a, v16h b, v8f c) {
  // (neg_a, A, neg_b, B, c_mod, C, reuse_a, reuse_b)
  return __builtin_amdgcn_wmma_f32_16x16x32_f16(false, a, false, b, (short)0, c, false, false);
}

// K index pattern for 16-bit A/B fragments (16x32), per CDNA5 ISA 7.12.2:
// lanes 0-15 hold K {0..7,16..23} pairs, lanes 16-31 hold K {8..15,24..31}.
__device__ __forceinline__ int frag_k(int j, int lane_hi) {
  int kh = lane_hi << 3;
  return (j < 4) ? (kh + (j << 1)) : (16 + kh + ((j - 4) << 1));
}

// f16 fragment gather (no conversion); p==nullptr or k>=K -> 0.
__device__ __forceinline__ v16h load_frag_h(const _Float16* __restrict__ p, int K,
                                            int k0, int lane_hi) {
  v16h r;
#pragma unroll
  for (int j = 0; j < 8; ++j) {
    int k = k0 + frag_k(j, lane_hi);
    h2 v; v.x = (_Float16)0.f; v.y = (_Float16)0.f;
    if (p != nullptr && k < K) v = *(const h2*)(p + k);
    r[2 * j]     = v.x;
    r[2 * j + 1] = v.y;
  }
  return r;
}

// Concatenated-K f16 fragment: k<K0 from p0, K0<=k<Ktot from p1 (even split).
__device__ __forceinline__ v16h load_frag_h_cat(const _Float16* __restrict__ p0, int K0,
                                                const _Float16* __restrict__ p1, int Ktot,
                                                int k0, int lane_hi) {
  v16h r;
#pragma unroll
  for (int j = 0; j < 8; ++j) {
    int k = k0 + frag_k(j, lane_hi);
    h2 v; v.x = (_Float16)0.f; v.y = (_Float16)0.f;
    if (k < K0)        v = *(const h2*)(p0 + k);
    else if (k < Ktot) v = *(const h2*)(p1 + (k - K0));
    r[2 * j]     = v.x;
    r[2 * j + 1] = v.y;
  }
  return r;
}

__device__ __forceinline__ float softplusf(float x) {
  return fmaxf(x, 0.f) + log1pf(expf(-fabsf(x)));
}
__device__ __forceinline__ float sigmoidf_(float x) { return 1.f / (1.f + expf(-x)); }

__global__ void cvt_f16_kernel(const float* __restrict__ src, _Float16* __restrict__ dst,
                               int n) {
  int i = blockIdx.x * blockDim.x + threadIdx.x;
  if (i < n) dst[i] = (_Float16)src[i];
}

__global__ void zero_h_kernel(_Float16* p, int n) {
  int i = blockIdx.x * blockDim.x + threadIdx.x;
  if (i < n) p[i] = (_Float16)0.f;
}

// ---------------- backward RNN step (fused input projection) -----------------
// hout[b][n] = tanh( x_t[b]·Wih[n] + hprev[b]·Whh[n] + bih[n] + bhh[n] )
// GEMM: M=128, N=400, K=88+400 (concatenated), all-f16 operands.
__global__ void rnn_step_kernel(const _Float16* __restrict__ x_t,  // x16 + t*IN
                                const _Float16* __restrict__ Wih,
                                const _Float16* __restrict__ Whh,
                                const float* __restrict__ bih,
                                const float* __restrict__ bhh,
                                const _Float16* __restrict__ hprev,
                                _Float16* __restrict__ hout) {
  const int wave = threadIdx.x >> 5, lane = threadIdx.x & 31;
  const int NT = kH / 16;  // 25
  int tile = blockIdx.x * (blockDim.x >> 5) + wave;
  if (tile >= (kB / 16) * NT) return;
  int mt = tile / NT, nt = tile % NT;
  int m0 = mt * 16, n0 = nt * 16;
  int lane_lo = lane & 15, lane_hi = lane >> 4;
  int mrow = m0 + lane_lo;            // batch row, always < 128
  int ncol = n0 + lane_lo;            // output column, always < 400
  const _Float16* xrow = x_t + mrow * (kS * kIN);
  const _Float16* hrow = hprev + mrow * kH;
  const _Float16* wi = Wih + ncol * kIN;
  const _Float16* wh = Whh + ncol * kH;
  const int Ktot = kIN + kH;          // 488
  v8f acc = {};
  for (int k0 = 0; k0 < Ktot; k0 += 32) {
    v16h a = load_frag_h_cat(xrow, kIN, hrow, Ktot, k0, lane_hi);
    v16h b = load_frag_h_cat(wi,   kIN, wh,   Ktot, k0, lane_hi);
    acc = wmma_f16(a, b, acc);
  }
#pragma unroll
  for (int r = 0; r < 8; ++r) {
    int m = m0 + r + (lane_hi << 3);
    int n = n0 + lane_lo;
    hout[m * kH + n] = (_Float16)tanhf(acc[r] + bih[n] + bhh[n]);
  }
}

// ---------------- combiner: h = 0.5*(tanh(z@Wc.T+bc) + hR[t]) ----------------
__global__ void comb_h_kernel(const _Float16* __restrict__ zprev,
                              const _Float16* __restrict__ Wc,
                              const float* __restrict__ bc,
                              const _Float16* __restrict__ hR_t,
                              _Float16* __restrict__ hbuf) {
  const int wave = threadIdx.x >> 5, lane = threadIdx.x & 31;
  const int NT = kH / 16;
  int tile = blockIdx.x * (blockDim.x >> 5) + wave;
  if (tile >= (kB / 16) * NT) return;
  int mt = tile / NT, nt = tile % NT;
  int m0 = mt * 16, n0 = nt * 16;
  int lane_lo = lane & 15, lane_hi = lane >> 4;
  const _Float16* arow = zprev + (m0 + lane_lo) * kL;
  const _Float16* brow = Wc + (n0 + lane_lo) * kL;
  v8f acc = {};
  for (int k0 = 0; k0 < kL; k0 += 32) {
    v16h a = load_frag_h(arow, kL, k0, lane_hi);
    v16h b = load_frag_h(brow, kL, k0, lane_hi);
    acc = wmma_f16(a, b, acc);
  }
#pragma unroll
  for (int r = 0; r < 8; ++r) {
    int m = m0 + r + (lane_hi << 3);
    int n = n0 + lane_lo;
    float hr = (float)hR_t[m * kH + n];
    hbuf[m * kH + n] = (_Float16)(0.5f * (tanhf(acc[r] + bc[n]) + hr));
  }
}

// -------- combiner: mu/var share the A fragment (two B matrices) -------------
__global__ void comb_muvar_kernel(const _Float16* __restrict__ hbuf,
                                  const _Float16* __restrict__ Wmu,
                                  const float* __restrict__ bmu,
                                  const _Float16* __restrict__ Wsig,
                                  const float* __restrict__ bsig,
                                  const float* __restrict__ noise_t,
                                  float* __restrict__ o_mui, float* __restrict__ o_sgi,
                                  _Float16* __restrict__ zbuf,
                                  _Float16* __restrict__ zprev, int t) {
  const int wave = threadIdx.x >> 5, lane = threadIdx.x & 31;
  const int NT = (kL + 15) / 16;  // 7
  int tile = blockIdx.x * (blockDim.x >> 5) + wave;
  if (tile >= (kB / 16) * NT) return;
  int mt = tile / NT, nt = tile % NT;
  int m0 = mt * 16, n0 = nt * 16;
  int lane_lo = lane & 15, lane_hi = lane >> 4;
  int ncol = n0 + lane_lo;
  const _Float16* arow  = hbuf + (m0 + lane_lo) * kH;
  const _Float16* brow0 = (ncol < kL) ? (Wmu  + ncol * kH) : nullptr;
  const _Float16* brow1 = (ncol < kL) ? (Wsig + ncol * kH) : nullptr;
  v8f accm = {}, accs = {};
  for (int k0 = 0; k0 < kH; k0 += 32) {
    v16h a  = load_frag_h(arow, kH, k0, lane_hi);
    v16h b0 = load_frag_h(brow0, kH, k0, lane_hi);
    v16h b1 = load_frag_h(brow1, kH, k0, lane_hi);
    accm = wmma_f16(a, b0, accm);
    accs = wmma_f16(a, b1, accs);
  }
#pragma unroll
  for (int r = 0; r < 8; ++r) {
    int m = m0 + r + (lane_hi << 3);
    int n = n0 + lane_lo;
    if (n < kL) {
      float mu  = accm[r] + bmu[n];
      float var = softplusf(accs[r] + bsig[n]);
      float z   = mu + sqrtf(var) * noise_t[m * kL + n];
      o_mui[m * kS * kL + t * kL + n] = mu;
      o_sgi[m * kS * kL + t * kL + n] = var;
      zbuf[(m * kS + t) * kL + n]     = (_Float16)z;
      zprev[m * kL + n]               = (_Float16)z;
    }
  }
}

// ---------------- decoder: generic dense + activation ------------------------
// out[m][n] = act( A[m]·W[n] + b[n] ); act 0 = relu, 1 = sigmoid.
template <typename OutT>
__global__ void dense_kernel(const _Float16* __restrict__ A, int K,
                             const _Float16* __restrict__ W,
                             const float* __restrict__ bias,
                             int N, OutT* __restrict__ out, int act, int Mtot) {
  const int wave = threadIdx.x >> 5, lane = threadIdx.x & 31;
  const int NT = (N + 15) / 16;
  int tile = blockIdx.x * (blockDim.x >> 5) + wave;
  if (tile >= (Mtot / 16) * NT) return;
  int mt = tile / NT, nt = tile % NT;
  int m0 = mt * 16, n0 = nt * 16;
  int lane_lo = lane & 15, lane_hi = lane >> 4;
  int ncol = n0 + lane_lo;
  const _Float16* arow = A + (m0 + lane_lo) * K;
  const _Float16* brow = (ncol < N) ? (W + ncol * K) : nullptr;
  v8f acc = {};
  for (int k0 = 0; k0 < K; k0 += 32) {
    v16h a = load_frag_h(arow, K, k0, lane_hi);
    v16h b = load_frag_h(brow, K, k0, lane_hi);
    acc = wmma_f16(a, b, acc);
  }
#pragma unroll
  for (int r = 0; r < 8; ++r) {
    int m = m0 + r + (lane_hi << 3);
    int n = n0 + lane_lo;
    if (n < N) {
      float v = acc[r] + bias[n];
      v = (act == 0) ? fmaxf(v, 0.f) : sigmoidf_(v);
      out[m * N + n] = (OutT)v;
    }
  }
}

// -------- generator stage A: a1=relu(z@Wg1.T), a2=relu(z@Wh1.T), zm=z@Wmg.T --
__global__ void gen_a_kernel(const _Float16* __restrict__ zg,
                             const _Float16* __restrict__ Wg1, const float* __restrict__ bg1,
                             const _Float16* __restrict__ Wh1, const float* __restrict__ bh1,
                             const _Float16* __restrict__ Wmg, const float* __restrict__ bmg,
                             _Float16* __restrict__ a1, _Float16* __restrict__ a2,
                             float* __restrict__ zm) {
  const int wave = threadIdx.x >> 5, lane = threadIdx.x & 31;
  int tile = blockIdx.x * (blockDim.x >> 5) + wave;
  if (tile >= 264) return;  // 104 (a1) + 104 (a2) + 56 (zm)
  int region, idx;
  if (tile < 104)      { region = 0; idx = tile; }
  else if (tile < 208) { region = 1; idx = tile - 104; }
  else                 { region = 2; idx = tile - 208; }
  const int NT = (region == 2) ? 7 : 13;
  const int N  = (region == 2) ? kL : kHTR;
  const _Float16* W = (region == 0) ? Wg1 : ((region == 1) ? Wh1 : Wmg);
  const float* bb   = (region == 0) ? bg1 : ((region == 1) ? bh1 : bmg);
  int mt = idx / NT, nt = idx % NT;
  int m0 = mt * 16, n0 = nt * 16;
  int lane_lo = lane & 15, lane_hi = lane >> 4;
  int ncol = n0 + lane_lo;
  const _Float16* arow = zg + (m0 + lane_lo) * kL;
  const _Float16* brow = (ncol < N) ? (W + ncol * kL) : nullptr;
  v8f acc = {};
  for (int k0 = 0; k0 < kL; k0 += 32) {
    v16h a = load_frag_h(arow, kL, k0, lane_hi);
    v16h b = load_frag_h(brow, kL, k0, lane_hi);
    acc = wmma_f16(a, b, acc);
  }
#pragma unroll
  for (int r = 0; r < 8; ++r) {
    int m = m0 + r + (lane_hi << 3);
    int n = n0 + lane_lo;
    if (n < N) {
      float v = acc[r] + bb[n];
      if (region == 0)      a1[m * N + n] = (_Float16)fmaxf(v, 0.f);
      else if (region == 1) a2[m * N + n] = (_Float16)fmaxf(v, 0.f);
      else                  zm[m * N + n] = v;
    }
  }
}

// -------- generator stage B: g=sigmoid(a1@Wg2.T), hp=a2@Wh2.T (shared tile) --
// Also emits relu(hp) in f16 for stage C's A-matrix.
__global__ void gen_b_kernel(const _Float16* __restrict__ a1,
                             const _Float16* __restrict__ a2,
                             const _Float16* __restrict__ Wg2, const float* __restrict__ bg2,
                             const _Float16* __restrict__ Wh2, const float* __restrict__ bh2,
                             float* __restrict__ g, float* __restrict__ hp,
                             _Float16* __restrict__ hp_relu) {
  const int wave = threadIdx.x >> 5, lane = threadIdx.x & 31;
  const int NT = (kL + 15) / 16;
  int tile = blockIdx.x * (blockDim.x >> 5) + wave;
  if (tile >= (kB / 16) * NT) return;
  int mt = tile / NT, nt = tile % NT;
  int m0 = mt * 16, n0 = nt * 16;
  int lane_lo = lane & 15, lane_hi = lane >> 4;
  int ncol = n0 + lane_lo;
  const _Float16* arow0 = a1 + (m0 + lane_lo) * kHTR;
  const _Float16* arow1 = a2 + (m0 + lane_lo) * kHTR;
  const _Float16* brow0 = (ncol < kL) ? (Wg2 + ncol * kHTR) : nullptr;
  const _Float16* brow1 = (ncol < kL) ? (Wh2 + ncol * kHTR) : nullptr;
  v8f accg = {}, acch = {};
  for (int k0 = 0; k0 < kHTR; k0 += 32) {
    v16h ag = load_frag_h(arow0, kHTR, k0, lane_hi);
    v16h bg = load_frag_h(brow0, kHTR, k0, lane_hi);
    accg = wmma_f16(ag, bg, accg);
    v16h ah = load_frag_h(arow1, kHTR, k0, lane_hi);
    v16h bh = load_frag_h(brow1, kHTR, k0, lane_hi);
    acch = wmma_f16(ah, bh, acch);
  }
#pragma unroll
  for (int r = 0; r < 8; ++r) {
    int m = m0 + r + (lane_hi << 3);
    int n = n0 + lane_lo;
    if (n < kL) {
      float hv = acch[r] + bh2[n];
      g[m * kL + n]       = sigmoidf_(accg[r] + bg2[n]);
      hp[m * kL + n]      = hv;
      hp_relu[m * kL + n] = (_Float16)fmaxf(hv, 0.f);
    }
  }
}

// -------- generator stage C: sigma GEMM + gated mix + reparam sample ---------
__global__ void gen_c_kernel(const _Float16* __restrict__ hp_relu,
                             const float* __restrict__ hp, const float* __restrict__ g,
                             const float* __restrict__ zm,
                             const _Float16* __restrict__ Wsg,
                             const float* __restrict__ bsg,
                             const float* __restrict__ noise_t,
                             float* __restrict__ o_mug, float* __restrict__ o_sgg,
                             _Float16* __restrict__ zg, int t) {
  const int wave = threadIdx.x >> 5, lane = threadIdx.x & 31;
  const int NT = (kL + 15) / 16;
  int tile = blockIdx.x * (blockDim.x >> 5) + wave;
  if (tile >= (kB / 16) * NT) return;
  int mt = tile / NT, nt = tile % NT;
  int m0 = mt * 16, n0 = nt * 16;
  int lane_lo = lane & 15, lane_hi = lane >> 4;
  int ncol = n0 + lane_lo;
  const _Float16* arow = hp_relu + (m0 + lane_lo) * kL;
  const _Float16* brow = (ncol < kL) ? (Wsg + ncol * kL) : nullptr;
  v8f acc = {};
  for (int k0 = 0; k0 < kL; k0 += 32) {
    v16h a = load_frag_h(arow, kL, k0, lane_hi);
    v16h b = load_frag_h(brow, kL, k0, lane_hi);
    acc = wmma_f16(a, b, acc);
  }
#pragma unroll
  for (int r = 0; r < 8; ++r) {
    int m = m0 + r + (lane_hi << 3);
    int n = n0 + lane_lo;
    if (n < kL) {
      float gv  = g[m * kL + n];
      float hv  = hp[m * kL + n];
      float zmv = zm[m * kL + n];
      float mu  = hv * gv + zmv * (1.f - gv);
      float sig = softplusf(acc[r] + bsg[n]);
      float z   = mu + sqrtf(sig) * noise_t[m * kL + n];
      o_mug[m * kS * kL + t * kL + n] = mu;
      o_sgg[m * kS * kL + t * kL + n] = sig;
      zg[m * kL + n] = (_Float16)z;
    }
  }
}

static inline void cvt(const float* s, _Float16* d, int n, hipStream_t st) {
  cvt_f16_kernel<<<(n + 255) / 256, 256, 0, st>>>(s, d, n);
}

extern "C" void kernel_launch(void* const* d_in, const int* in_sizes, int n_in,
                              void* d_out, int out_size, void* d_ws, size_t ws_size,
                              hipStream_t stream) {
  (void)in_sizes; (void)n_in; (void)out_size; (void)ws_size;
  const float* x      = (const float*)d_in[0];
  // d_in[1..4] = forward-RNN params: dead code in the reference, skipped.
  const float* Wih_b  = (const float*)d_in[5];
  const float* Whh_b  = (const float*)d_in[6];
  const float* bih_b  = (const float*)d_in[7];
  const float* bhh_b  = (const float*)d_in[8];
  const float* Wc     = (const float*)d_in[9];
  const float* bc     = (const float*)d_in[10];
  const float* Wmu    = (const float*)d_in[11];
  const float* bmu    = (const float*)d_in[12];
  const float* Wsig   = (const float*)d_in[13];
  const float* bsig   = (const float*)d_in[14];
  const float* We1    = (const float*)d_in[15];
  const float* be1    = (const float*)d_in[16];
  const float* We2    = (const float*)d_in[17];
  const float* be2    = (const float*)d_in[18];
  const float* We3    = (const float*)d_in[19];
  const float* be3    = (const float*)d_in[20];
  const float* Wg1    = (const float*)d_in[21];
  const float* bg1    = (const float*)d_in[22];
  const float* Wg2    = (const float*)d_in[23];
  const float* bg2    = (const float*)d_in[24];
  const float* Wh1    = (const float*)d_in[25];
  const float* bh1    = (const float*)d_in[26];
  const float* Wh2    = (const float*)d_in[27];
  const float* bh2    = (const float*)d_in[28];
  const float* Wmg    = (const float*)d_in[29];
  const float* bmg    = (const float*)d_in[30];
  const float* Wsg    = (const float*)d_in[31];
  const float* bsg    = (const float*)d_in[32];
  const float* noise_inf = (const float*)d_in[33];
  const float* z0_gen    = (const float*)d_in[34];
  const float* noise_gen = (const float*)d_in[35];

  // ---- workspace layout: small f32 elementwise buffers, then f16 region ----
  float* wsf  = (float*)d_ws;
  float* zm   = wsf;                 // B*L
  float* gbuf = zm + kB * kL;        // B*L
  float* hp   = gbuf + kB * kL;      // B*L
  _Float16* h16 = (_Float16*)(hp + kB * kL);
  _Float16* hR16    = h16;                        // S*B*H
  _Float16* x16     = hR16 + kS * kB * kH;        // B*S*IN
  _Float16* zbuf16  = x16 + kB * kS * kIN;        // B*S*L  (reused as e2)
  _Float16* e116    = zbuf16 + kB * kS * kL;      // B*S*HEM
  _Float16* hbuf16  = e116 + kB * kS * kHEM;      // B*H
  _Float16* zprev16 = hbuf16 + kB * kH;           // B*L
  _Float16* zg16    = zprev16 + kB * kL;          // B*L
  _Float16* a116    = zg16 + kB * kL;             // B*HTR
  _Float16* a216    = a116 + kB * kHTR;           // B*HTR
  _Float16* hpr16   = a216 + kB * kHTR;           // B*L  relu(h_prop)
  _Float16* z0g16   = hpr16 + kB * kL;            // B*L
  _Float16* hzero16 = z0g16 + kB * kL;            // B*H  zeros
  _Float16* zzero16 = hzero16 + kB * kH;          // B*L  zeros
  // f16 weights
  _Float16* Wih16  = zzero16 + kB * kL;           // H*IN
  _Float16* Whh16  = Wih16 + kH * kIN;            // H*H
  _Float16* Wc16   = Whh16 + kH * kH;             // H*L
  _Float16* Wmu16  = Wc16 + kH * kL;              // L*H
  _Float16* Wsi16  = Wmu16 + kL * kH;             // L*H (Wsig)
  _Float16* We116  = Wsi16 + kL * kH;             // HEM*L
  _Float16* We216  = We116 + kHEM * kL;           // HEM*HEM
  _Float16* We316  = We216 + kHEM * kHEM;         // IN*HEM
  _Float16* Wg116  = We316 + kIN * kHEM;          // HTR*L
  _Float16* Wg216  = Wg116 + kHTR * kL;           // L*HTR
  _Float16* Wh116  = Wg216 + kL * kHTR;           // HTR*L
  _Float16* Wh216  = Wh116 + kHTR * kL;           // L*HTR
  _Float16* Wmg16  = Wh216 + kL * kHTR;           // L*L
  _Float16* Wsg16  = Wmg16 + kL * kL;             // L*L

  float* out    = (float*)d_out;
  float* o_xhat = out;
  float* o_mui  = o_xhat + kB * kS * kIN;
  float* o_sgi  = o_mui  + kB * kS * kL;
  float* o_mug  = o_sgi  + kB * kS * kL;
  float* o_sgg  = o_mug  + kB * kS * kL;

  // ---- one-time (per launch) f32->f16 conversions; weights stay L2-hot ----
  cvt(x, x16, kB * kS * kIN, stream);
  cvt(z0_gen, z0g16, kB * kL, stream);
  cvt(Wih_b, Wih16, kH * kIN, stream);
  cvt(Whh_b, Whh16, kH * kH, stream);
  cvt(Wc,   Wc16,  kH * kL, stream);
  cvt(Wmu,  Wmu16, kL * kH, stream);
  cvt(Wsig, Wsi16, kL * kH, stream);
  cvt(We1,  We116, kHEM * kL, stream);
  cvt(We2,  We216, kHEM * kHEM, stream);
  cvt(We3,  We316, kIN * kHEM, stream);
  cvt(Wg1,  Wg116, kHTR * kL, stream);
  cvt(Wg2,  Wg216, kL * kHTR, stream);
  cvt(Wh1,  Wh116, kHTR * kL, stream);
  cvt(Wh2,  Wh216, kL * kHTR, stream);
  cvt(Wmg,  Wmg16, kL * kL, stream);
  cvt(Wsg,  Wsg16, kL * kL, stream);
  {
    int n = kB * kH + kB * kL;  // hzero16 .. zzero16 contiguous
    zero_h_kernel<<<(n + 255) / 256, 256, 0, stream>>>(hzero16, n);
  }

  // Backward RNN: t = S-1 .. 0, each step a fused 128x400x488 WMMA GEMM.
  for (int t = kS - 1; t >= 0; --t) {
    const _Float16* hprev = (t == kS - 1) ? hzero16 : (hR16 + (t + 1) * kB * kH);
    rnn_step_kernel<<<50, 128, 0, stream>>>(x16 + t * kIN, Wih16, Whh16, bih_b, bhh_b,
                                            hprev, hR16 + t * kB * kH);
  }

  // Combiner scan: t = 0 .. S-1.
  for (int t = 0; t < kS; ++t) {
    const _Float16* zp = (t == 0) ? zzero16 : zprev16;
    comb_h_kernel<<<50, 128, 0, stream>>>(zp, Wc16, bc, hR16 + t * kB * kH, hbuf16);
    comb_muvar_kernel<<<14, 128, 0, stream>>>(hbuf16, Wmu16, bmu, Wsi16, bsig,
                                              noise_inf + t * kB * kL,
                                              o_mui, o_sgi, zbuf16, zprev16, t);
  }

  // Decoder (fully parallel over B*S = 32768 rows).
  dense_kernel<_Float16><<<3584, 128, 0, stream>>>(zbuf16, kL,   We116, be1, kHEM, e116,   0, kB * kS);
  dense_kernel<_Float16><<<3584, 128, 0, stream>>>(e116,   kHEM, We216, be2, kHEM, zbuf16, 0, kB * kS);
  dense_kernel<float>   <<<3072, 128, 0, stream>>>(zbuf16, kHEM, We316, be3, kIN,  o_xhat, 1, kB * kS);

  // Generator scan: t = 0 .. S-1.
  for (int t = 0; t < kS; ++t) {
    const _Float16* z_in = (t == 0) ? z0g16 : zg16;
    gen_a_kernel<<<66, 128, 0, stream>>>(z_in, Wg116, bg1, Wh116, bh1, Wmg16, bmg,
                                         a116, a216, zm);
    gen_b_kernel<<<14, 128, 0, stream>>>(a116, a216, Wg216, bg2, Wh216, bh2,
                                         gbuf, hp, hpr16);
    gen_c_kernel<<<14, 128, 0, stream>>>(hpr16, hp, gbuf, zm, Wsg16, bsg,
                                         noise_gen + t * kB * kL,
                                         o_mug, o_sgg, zg16, t);
  }
}
